// WeightedCausalAttention_27006754357924
// MI455X (gfx1250) — matrix operand order repaired and checked by reading, not compiled, not run on hardware
//
#include <hip/hip_runtime.h>

typedef _Float16 v16h __attribute__((ext_vector_type(16)));
typedef float    v8f  __attribute__((ext_vector_type(8)));
typedef unsigned int u32x4 __attribute__((ext_vector_type(4)));

union Tile16 {
  v16h  v;
  u32x4 q[2];
};

#define WMMA16(a, b, c) \
  __builtin_amdgcn_wmma_f32_16x16x32_f16(false, (a), false, (b), (short)0, (c), false, false)

// Async copy 16B global -> LDS (CDNA5 GLOBAL_LOAD_ASYNC_TO_LDS_B128, ASYNCcnt)
__device__ __forceinline__ void async_stage16(_Float16* ldsDst, const _Float16* gsrc) {
  unsigned off = (unsigned)(unsigned long long)ldsDst;   // low 32 bits = LDS byte offset
  asm volatile("global_load_async_to_lds_b128 %0, %1, off"
               :: "v"(off), "v"((unsigned long long)gsrc)
               : "memory");
}
__device__ __forceinline__ void wait_async_le1() {
  asm volatile("s_wait_asynccnt 1" ::: "memory");
}

// ---------------------------------------------------------------------------
// Kernel A1: f32 -> f16 elementwise convert
// ---------------------------------------------------------------------------
__global__ void cvt_f32_f16(const float* __restrict__ src, _Float16* __restrict__ dst, int n) {
  int i = blockIdx.x * blockDim.x + threadIdx.x;
  int stride = gridDim.x * blockDim.x;
  for (; i < n; i += stride) dst[i] = (_Float16)src[i];
}

// ---------------------------------------------------------------------------
// Kernel A2: f32 -> f16 transpose convert: Wt[n][k] = W[k][n], dim x dim
// ---------------------------------------------------------------------------
__global__ void cvt_transpose(const float* __restrict__ W, _Float16* __restrict__ Wt, int dim) {
  int i = blockIdx.x * blockDim.x + threadIdx.x;
  int total = dim * dim;
  int stride = gridDim.x * blockDim.x;
  for (; i < total; i += stride) {
    int k = i / dim, n = i - k * dim;
    Wt[n * dim + k] = (_Float16)W[i];
  }
}

// ---------------------------------------------------------------------------
// Shared staged K-loop: block tile 128(M) x 64(N), 8 waves (one 16-row strip
// each), B panel (64 rows x 32 k, f16 = 4KB) double-buffered in LDS via
// async-to-LDS copies. All 256 threads must participate (uniform barriers).
// Per step: one wait, then 4 back-to-back WMMAs (operands batched first).
// ---------------------------------------------------------------------------
__device__ __forceinline__ void staged_kloop(const _Float16* __restrict__ arow,
                                             const _Float16* __restrict__ Wpanel,
                                             _Float16* bsh,
                                             int tid, int lo, int hi,
                                             v8f acc[4]) {
  const int srow = tid >> 2;          // 0..63 : weight row within panel
  const int skc  = (tid & 3) * 8;     // halfword chunk within 32-k tile
  const _Float16* gbase = Wpanel + (size_t)srow * 1024 + skc;
  const int lofs = srow * 32 + skc;   // LDS halfword offset within buffer

  // prologue: stage k=0 into buffer 0
  async_stage16(bsh + lofs, gbase);

  for (int k0 = 0; k0 < 1024; k0 += 32) {
    int cur = (k0 >> 5) & 1;
    int nxt = cur ^ 1;
    __syncthreads();                                    // readers of buf[nxt] done
    async_stage16(bsh + nxt * 2048 + lofs, gbase + ((k0 + 32) & 1023));
    wait_async_le1();                                   // own buf[cur] load done
    __syncthreads();                                    // everyone's buf[cur] done

    // batch all operand loads, then issue 4 WMMAs back-to-back
    Tile16 a, bt[4];
    a.q[0] = *(const u32x4*)(arow + k0 + 16 * hi);
    a.q[1] = *(const u32x4*)(arow + k0 + 16 * hi + 8);
    const _Float16* bcur = bsh + cur * 2048;
#pragma unroll
    for (int nt = 0; nt < 4; nt++) {
      const _Float16* brow = bcur + (nt * 16 + lo) * 32 + 16 * hi;
      bt[nt].q[0] = *(const u32x4*)(brow);
      bt[nt].q[1] = *(const u32x4*)(brow + 8);
    }
#pragma unroll
    for (int nt = 0; nt < 4; nt++) {
      acc[nt] = WMMA16(a.v, bt[nt].v, acc[nt]);
    }
  }
}

// ---------------------------------------------------------------------------
// Kernel B: fused QKV projection GEMM.
// grid = 3 mats * 32 mBlocks * 16 nStrips = 1536 blocks of 256 threads.
//   Out: Qb [B,H,N,dk] (pre-scaled 1/8), Kb [B,H,N,dk], Vt [B,H,dk,N]
// ---------------------------------------------------------------------------
__global__ void qkv_gemm(const _Float16* __restrict__ xb,
                         const _Float16* __restrict__ Wqt,
                         const _Float16* __restrict__ Wkt,
                         const _Float16* __restrict__ Wvt,
                         const float* __restrict__ bq,
                         const float* __restrict__ bk,
                         const float* __restrict__ bv,
                         _Float16* __restrict__ Qb,
                         _Float16* __restrict__ Kb,
                         _Float16* __restrict__ Vt) {
  __shared__ __align__(16) _Float16 bsh[2 * 64 * 32];
  const int D = 1024;
  int tid = threadIdx.x;
  int wv = tid >> 5;
  int lane = tid & 31;
  int lo = lane & 15, hi = lane >> 4;

  int mat = blockIdx.x / (32 * 16);
  int rem = blockIdx.x - mat * (32 * 16);
  int mBase = (rem >> 4) * 128 + wv * 16;
  int nBase = (rem & 15) * 64;

  const _Float16* Wt   = (mat == 0) ? Wqt : ((mat == 1) ? Wkt : Wvt);
  const float*    bias = (mat == 0) ? bq  : ((mat == 1) ? bk  : bv);

  v8f acc[4] = {};
  const _Float16* arow = xb + (size_t)(mBase + lo) * D;
  staged_kloop(arow, Wt + (size_t)nBase * D, bsh, tid, lo, hi, acc);

#pragma unroll
  for (int nt = 0; nt < 4; nt++) {
    int ncol = nBase + nt * 16 + lo;
    float bsv = bias[ncol];
    int h = ncol >> 6, d = ncol & 63;
#pragma unroll
    for (int r = 0; r < 8; r++) {
      int m = mBase + r + 8 * hi;
      int bb = m >> 11, nrow = m & 2047;
      float val = acc[nt][r] + bsv;
      if (mat == 0) {
        Qb[(size_t)(((bb * 16 + h) * 2048) + nrow) * 64 + d] = (_Float16)(val * 0.125f);
      } else if (mat == 1) {
        Kb[(size_t)(((bb * 16 + h) * 2048) + nrow) * 64 + d] = (_Float16)val;
      } else {
        Vt[(size_t)(((bb * 16 + h) * 64) + d) * 2048 + nrow] = (_Float16)val;
      }
    }
  }
}

// ---------------------------------------------------------------------------
// Kernel C: flash attention with power-law decay mask.
// 1 wave (blockDim=32) per 16-row query block per (b,h). Key blocks of 32.
// grid = B*H*(N/16) = 4096 blocks.
// ---------------------------------------------------------------------------
__global__ void attn_kernel(const _Float16* __restrict__ Qb,
                            const _Float16* __restrict__ Kb,
                            const _Float16* __restrict__ Vt,
                            _Float16* __restrict__ Ob) {
  const int N = 2048, DK = 64;
  __shared__ __align__(16) _Float16 lds[16 * 40];

  int bh = blockIdx.x >> 7;       // 0..31
  int mtile = blockIdx.x & 127;
  int mBase = mtile * 16;
  int lane = threadIdx.x & 31;
  int lo = lane & 15, hi = lane >> 4;

  const _Float16* Qh = Qb + (size_t)bh * N * DK;
  const _Float16* Kh = Kb + (size_t)bh * N * DK;
  const _Float16* Vh = Vt + (size_t)bh * DK * N;

  // Preload Q A-tiles (k = 0..31 and 32..63)
  Tile16 qa0, qa1;
  {
    const _Float16* qrow = Qh + (size_t)(mBase + lo) * DK;
    qa0.q[0] = *(const u32x4*)(qrow + 16 * hi);
    qa0.q[1] = *(const u32x4*)(qrow + 16 * hi + 8);
    qa1.q[0] = *(const u32x4*)(qrow + 32 + 16 * hi);
    qa1.q[1] = *(const u32x4*)(qrow + 32 + 16 * hi + 8);
  }

  v8f acc[4] = {};
  float mstat[8], lstat[8];
#pragma unroll
  for (int r = 0; r < 8; r++) { mstat[r] = -__builtin_inff(); lstat[r] = 0.0f; }

  int jbmax = (mBase + 15) >> 5;
  for (int jb = 0; jb <= jbmax; jb++) {
    int jBase = jb * 32;

    // Prefetch next key/value block one iteration ahead (global_prefetch_b8)
    if (jb < jbmax) {
      __builtin_prefetch(Kh + (size_t)(jBase + 32 + lane) * DK, 0, 3);
      __builtin_prefetch(Vh + (size_t)(lane << 1) * N + jBase + 32, 0, 3);
    }

    // Batch the 4 K B-tiles, then 4 back-to-back WMMAs for S = Q K^T
    Tile16 kb[4];
    {
      const _Float16* krow0 = Kh + (size_t)(jBase + lo) * DK;
      const _Float16* krow1 = Kh + (size_t)(jBase + 16 + lo) * DK;
      kb[0].q[0] = *(const u32x4*)(krow0 + 16 * hi);
      kb[0].q[1] = *(const u32x4*)(krow0 + 16 * hi + 8);
      kb[1].q[0] = *(const u32x4*)(krow0 + 32 + 16 * hi);
      kb[1].q[1] = *(const u32x4*)(krow0 + 32 + 16 * hi + 8);
      kb[2].q[0] = *(const u32x4*)(krow1 + 16 * hi);
      kb[2].q[1] = *(const u32x4*)(krow1 + 16 * hi + 8);
      kb[3].q[0] = *(const u32x4*)(krow1 + 32 + 16 * hi);
      kb[3].q[1] = *(const u32x4*)(krow1 + 32 + 16 * hi + 8);
    }
    v8f s0 = {}, s1 = {};
    s0 = WMMA16(qa0.v, kb[0].v, s0);
    s0 = WMMA16(qa1.v, kb[1].v, s0);
    s1 = WMMA16(qa0.v, kb[2].v, s1);
    s1 = WMMA16(qa1.v, kb[3].v, s1);

    // Issue V-tile loads early: latency hides under the softmax VALU work.
    Tile16 vb[4];
#pragma unroll
    for (int nt = 0; nt < 4; nt++) {
      const _Float16* vrow = Vh + (size_t)(nt * 16 + lo) * N + jBase + 16 * hi;
      vb[nt].q[0] = *(const u32x4*)(vrow);
      vb[nt].q[1] = *(const u32x4*)(vrow + 8);
    }

    // Power-law causal mask + online softmax
    float p0[8], p1[8];
#pragma unroll
    for (int r = 0; r < 8; r++) {
      int m = mBase + r + 8 * hi;
      int c0 = jBase + lo, c1 = jBase + 16 + lo;
      float v0 = s0[r], v1 = s1[r];
      int d0 = m - c0, d1 = m - c1;
      v0 = (d0 < 0) ? -__builtin_inff() : ((d0 > 0) ? v0 - 0.25f * logf((float)d0) : v0);
      v1 = (d1 < 0) ? -__builtin_inff() : ((d1 > 0) ? v1 - 0.25f * logf((float)d1) : v1);

      float mx = fmaxf(v0, v1);
      mx = fmaxf(mx, __shfl_xor(mx, 1, 32));
      mx = fmaxf(mx, __shfl_xor(mx, 2, 32));
      mx = fmaxf(mx, __shfl_xor(mx, 4, 32));
      mx = fmaxf(mx, __shfl_xor(mx, 8, 32));

      float mnew  = fmaxf(mstat[r], mx);
      float alpha = expf(mstat[r] - mnew);
      float e0 = expf(v0 - mnew);
      float e1 = expf(v1 - mnew);
      float rs = e0 + e1;
      rs += __shfl_xor(rs, 1, 32);
      rs += __shfl_xor(rs, 2, 32);
      rs += __shfl_xor(rs, 4, 32);
      rs += __shfl_xor(rs, 8, 32);

      lstat[r] = lstat[r] * alpha + rs;
      mstat[r] = mnew;
#pragma unroll
      for (int nt = 0; nt < 4; nt++) acc[nt][r] *= alpha;
      p0[r] = e0;
      p1[r] = e1;
    }

    // P: C-layout -> LDS -> A-layout
#pragma unroll
    for (int r = 0; r < 8; r++) {
      lds[(r + 8 * hi) * 40 + lo]      = (_Float16)p0[r];
      lds[(r + 8 * hi) * 40 + 16 + lo] = (_Float16)p1[r];
    }
    __syncthreads();
    Tile16 pa;
    pa.q[0] = *(const u32x4*)(&lds[lo * 40 + 16 * hi]);
    pa.q[1] = *(const u32x4*)(&lds[lo * 40 + 16 * hi + 8]);
    __syncthreads();

    // O += P V  (V stored transposed: Vt[d][n]); vb already resident
#pragma unroll
    for (int nt = 0; nt < 4; nt++) {
      acc[nt] = WMMA16(pa.v, vb[nt].v, acc[nt]);
    }
  }

  // Normalize and write O as [B*N, 1024] f16 (row g, col h*64+d)
  int b = bh >> 4, h = bh & 15;
#pragma unroll
  for (int r = 0; r < 8; r++) {
    float inv = 1.0f / lstat[r];
    size_t grow = (size_t)(b * 2048 + mBase + r + 8 * hi);
#pragma unroll
    for (int nt = 0; nt < 4; nt++) {
      Ob[grow * 1024 + h * 64 + nt * 16 + lo] = (_Float16)(acc[nt][r] * inv);
    }
  }
}

// ---------------------------------------------------------------------------
// Kernel D: output projection. out = Ob @ Wo + bo  (f32 output)
// grid = 32 mBlocks * 16 nStrips = 512 blocks of 256 threads.
// ---------------------------------------------------------------------------
__global__ void out_gemm(const _Float16* __restrict__ Ob,
                         const _Float16* __restrict__ Wot,
                         const float* __restrict__ bo,
                         float* __restrict__ out) {
  __shared__ __align__(16) _Float16 bsh[2 * 64 * 32];
  const int D = 1024;
  int tid = threadIdx.x;
  int wv = tid >> 5;
  int lane = tid & 31;
  int lo = lane & 15, hi = lane >> 4;
  int mBase = (blockIdx.x >> 4) * 128 + wv * 16;
  int nBase = (blockIdx.x & 15) * 64;

  v8f acc[4] = {};
  const _Float16* arow = Ob + (size_t)(mBase + lo) * D;
  staged_kloop(arow, Wot + (size_t)nBase * D, bsh, tid, lo, hi, acc);

#pragma unroll
  for (int nt = 0; nt < 4; nt++) {
    int ncol = nBase + nt * 16 + lo;
    float bsv = bo[ncol];
#pragma unroll
    for (int r = 0; r < 8; r++) {
      out[(size_t)(mBase + r + 8 * hi) * D + ncol] = acc[nt][r] + bsv;
    }
  }
}

// ---------------------------------------------------------------------------
// Host launcher
// ---------------------------------------------------------------------------
extern "C" void kernel_launch(void* const* d_in, const int* in_sizes, int n_in,
                              void* d_out, int out_size, void* d_ws, size_t ws_size,
                              hipStream_t stream) {
  const float* x  = (const float*)d_in[0];
  const float* Wq = (const float*)d_in[1];
  const float* bq = (const float*)d_in[2];
  const float* Wk = (const float*)d_in[3];
  const float* bk = (const float*)d_in[4];
  const float* Wv = (const float*)d_in[5];
  const float* bv = (const float*)d_in[6];
  const float* Wo = (const float*)d_in[7];
  const float* bo = (const float*)d_in[8];
  float* out = (float*)d_out;

  char* ws = (char*)d_ws;
  const size_t MB = 1024u * 1024u;
  _Float16* xb  = (_Float16*)(ws + 0 * MB);    // 8 MB  : [4096 x 1024]
  _Float16* Wqt = (_Float16*)(ws + 8 * MB);    // 2 MB
  _Float16* Wkt = (_Float16*)(ws + 10 * MB);   // 2 MB
  _Float16* Wvt = (_Float16*)(ws + 12 * MB);   // 2 MB
  _Float16* Wot = (_Float16*)(ws + 14 * MB);   // 2 MB
  _Float16* Qb  = (_Float16*)(ws + 16 * MB);   // 8 MB  : [B,H,N,dk]
  _Float16* Kb  = (_Float16*)(ws + 24 * MB);   // 8 MB  : [B,H,N,dk]
  _Float16* Vt  = (_Float16*)(ws + 32 * MB);   // 8 MB  : [B,H,dk,N]
  _Float16* Ob  = (_Float16*)(ws + 40 * MB);   // 8 MB  : [B*N, 1024]

  cvt_f32_f16<<<1024, 256, 0, stream>>>(x, xb, 4096 * 1024);
  cvt_transpose<<<1024, 256, 0, stream>>>(Wq, Wqt, 1024);
  cvt_transpose<<<1024, 256, 0, stream>>>(Wk, Wkt, 1024);
  cvt_transpose<<<1024, 256, 0, stream>>>(Wv, Wvt, 1024);
  cvt_transpose<<<1024, 256, 0, stream>>>(Wo, Wot, 1024);

  // 3 mats * 32 mBlocks * 16 nStrips = 1536 blocks
  qkv_gemm<<<1536, 256, 0, stream>>>(xb, Wqt, Wkt, Wvt, bq, bk, bv, Qb, Kb, Vt);

  // B*H*(N/16) = 4096 single-wave blocks
  attn_kernel<<<4096, 32, 0, stream>>>(Qb, Kb, Vt, Ob);

  // 32 mBlocks * 16 nStrips = 512 blocks
  out_gemm<<<512, 256, 0, stream>>>(Ob, Wot, bo, out);
}
